// RelativeMultiheadAttention_38293928411465
// MI455X (gfx1250) — matrix operand order repaired and checked by reading, not compiled, not run on hardware
//
#include <hip/hip_runtime.h>
#include <hip/hip_bf16.h>

typedef __bf16 bf16;
typedef bf16 v16bf __attribute__((ext_vector_type(16)));
typedef bf16 v8bf  __attribute__((ext_vector_type(8)));
typedef bf16 v4bf  __attribute__((ext_vector_type(4)));
typedef float v8f  __attribute__((ext_vector_type(8)));
typedef int  v4i   __attribute__((ext_vector_type(4)));

typedef __attribute__((address_space(1))) v4i g_v4i;   // global int4
typedef __attribute__((address_space(3))) v4i l_v4i;   // LDS int4

#define EMBED   1024
#define HEADS   16
#define HD      64
#define SEQ     1024
#define BATCH   4
#define MAXLEN  2048

// ---------- async global->LDS copy (16 bytes), ASYNCcnt-tracked ----------
__device__ __forceinline__ void async_cp16(const void* g, void* l) {
#if __has_builtin(__builtin_amdgcn_global_load_async_to_lds_b128)
    __builtin_amdgcn_global_load_async_to_lds_b128(
        (g_v4i*)(uintptr_t)g, (l_v4i*)(uintptr_t)l, 0, 0);
#else
    const unsigned loff = (unsigned)(uintptr_t)l;
    asm volatile("global_load_async_to_lds_b128 %0, %1, off"
                 :: "v"(loff), "v"(g) : "memory");
#endif
}
__device__ __forceinline__ void wait_async0() {
#if __has_builtin(__builtin_amdgcn_s_wait_asynccnt)
    __builtin_amdgcn_s_wait_asynccnt(0);
#else
    asm volatile("s_wait_asynccnt 0" ::: "memory");
#endif
}

// ---------- WMMA fragment load helpers (bf16, 16x16x32) ----------
// A-matrix 16x32: lane<16 -> row M=lane,    K = 0..15
//                 lane>=16 -> row M=lane-16, K = 16..31.
// p points at element K = kstep + (lane>>4)*16 of the row.
__device__ __forceinline__ v16bf load_a_frag(const bf16* p) {
    v8bf lo = *(const v8bf*)p;
    v8bf hi = *(const v8bf*)(p + 8);
    v16bf r;
#pragma unroll
    for (int i = 0; i < 8; ++i) { r[i] = lo[i]; r[i + 8] = hi[i]; }
    return r;
}
// B-matrix 32x16: lane holds one column (N = lane&15), 16 consecutive K
// values starting at kstep + (lane>>4)*16.  p points at that element.
__device__ __forceinline__ v16bf load_b_frag(const bf16* p) {
    v8bf lo = *(const v8bf*)p;
    v8bf hi = *(const v8bf*)(p + 8);
    v16bf r;
#pragma unroll
    for (int i = 0; i < 8; ++i) { r[i] = lo[i]; r[i + 8] = hi[i]; }
    return r;
}

__device__ __forceinline__ v8f wmma_bf16(v16bf a, v16bf b, v8f c) {
    return __builtin_amdgcn_wmma_f32_16x16x32_bf16(
        false, a, false, b, (short)0, c, false, false);
}

// ---------- LDS staging: 32 contiguous elements per call ----------
__device__ __forceinline__ void stage32(const float* src, bf16* dst) {
#pragma unroll
    for (int i = 0; i < 8; ++i) {
        float4 f = *(const float4*)(src + 4 * i);
        v4bf o;
        o[0] = (bf16)f.x; o[1] = (bf16)f.y; o[2] = (bf16)f.z; o[3] = (bf16)f.w;
        *(v4bf*)(dst + 4 * i) = o;
    }
}

// =================================================================
// GEMM:  out = A(MxK) @ W(NxK)^T + bias
// MODE 0: out bf16, head-major [B,H,S,D]
// MODE 1: same as 0 plus rel_pos[h, s, d] added (K projection)
// MODE 2: out fp32, plain row-major [M,N] (final O projection)
// Block tile 128x128, 256 threads (8 waves), k-chunk 64.
// bf16 A tiles (MODE 2) are staged with async global->LDS DMA.
// =================================================================
template <typename AT, int MODE>
__global__ __launch_bounds__(256) void gemm_xwT(
    const AT* __restrict__ A, const float* __restrict__ W,
    const float* __restrict__ bias, const float* __restrict__ rel,
    void* __restrict__ outp)
{
    constexpr int K = EMBED, N = EMBED;
    __shared__ bf16 As[128][72];
    __shared__ bf16 Bs[128][72];

    const int tid  = threadIdx.x;
    const int wave = tid >> 5;
    const int lane = tid & 31;
    const int hi   = lane >> 4;
    const int ln   = lane & 15;

    const int bm = blockIdx.y * 128;
    const int bn = blockIdx.x * 128;
    const int wm = (wave >> 1) * 32;  // wave M offset (0,32,64,96)
    const int wn = (wave & 1) * 64;   // wave N offset (0,64)

    const int lr = tid >> 1;          // staging row 0..127
    const int lc = (tid & 1) * 32;    // staging col 0 or 32

    v8f acc[2][4];
#pragma unroll
    for (int i = 0; i < 2; ++i)
#pragma unroll
        for (int j = 0; j < 4; ++j) acc[i][j] = (v8f){};

    for (int kk = 0; kk < K; kk += 64) {
        __syncthreads();
        if constexpr (sizeof(AT) == 2) {
            // bf16 A tile: async DMA straight into LDS (no conversion needed)
            const bf16* src = (const bf16*)A + (size_t)(bm + lr) * K + kk + lc;
            bf16* dst = &As[lr][lc];
            async_cp16(src,      dst);
            async_cp16(src + 8,  dst + 8);
            async_cp16(src + 16, dst + 16);
            async_cp16(src + 24, dst + 24);
        } else {
            stage32((const float*)A + (size_t)(bm + lr) * K + kk + lc, &As[lr][lc]);
        }
        stage32(W + (size_t)(bn + lr) * K + kk + lc, &Bs[lr][lc]);
        if (kk + 64 < K) {
            __builtin_prefetch(W + (size_t)(bn + lr) * K + kk + 64 + lc, 0, 0);
        }
        if constexpr (sizeof(AT) == 2) wait_async0();
        __syncthreads();

#pragma unroll
        for (int ks = 0; ks < 64; ks += 32) {
            v16bf af[2], bf_[4];
#pragma unroll
            for (int i = 0; i < 2; ++i)
                af[i] = load_a_frag(&As[wm + i * 16 + ln][ks + hi * 16]);
#pragma unroll
            for (int j = 0; j < 4; ++j)
                bf_[j] = load_b_frag(&Bs[wn + j * 16 + ln][ks + hi * 16]);
#pragma unroll
            for (int i = 0; i < 2; ++i)
#pragma unroll
                for (int j = 0; j < 4; ++j)
                    acc[i][j] = wmma_bf16(af[i], bf_[j], acc[i][j]);
        }
    }

    // ---------------- epilogue ----------------
#pragma unroll
    for (int i = 0; i < 2; ++i) {
#pragma unroll
        for (int j = 0; j < 4; ++j) {
            const int n = bn + wn + j * 16 + ln;
            const float bsv = bias[n];
#pragma unroll
            for (int r = 0; r < 8; ++r) {
                const int m = bm + wm + i * 16 + r + hi * 8;
                float v = acc[i][j][r] + bsv;
                if (MODE == 2) {
                    ((float*)outp)[(size_t)m * N + n] = v;
                } else {
                    const int b = m >> 10, s = m & 1023;
                    const int h = n >> 6,  d = n & 63;
                    if (MODE == 1)
                        v += rel[((size_t)h * MAXLEN + s) * HD + d];
                    ((bf16*)outp)[((((size_t)b * HEADS + h) << 10) + s) * HD + d] =
                        (bf16)v;
                }
            }
        }
    }
}

// =================================================================
// Flash attention over head-major bf16 Q, K' (=K+rel), V.
// Block: 256 threads = 8 waves; each wave owns 16 q rows.
// grid.x = T/128, grid.y = B*H.  S-chunks of 64 keys.
// K chunks staged via async global->LDS DMA; V transposed through VGPRs.
// No 1/sqrt(d) scale (reference applies none).
// =================================================================
__global__ __launch_bounds__(256) void flash_attn(
    const bf16* __restrict__ Qh, const bf16* __restrict__ Kh,
    const bf16* __restrict__ Vh, const unsigned char* __restrict__ kpm,
    bf16* __restrict__ attn)
{
    __shared__ bf16 Ks[64][72];       // [key][d]
    __shared__ bf16 Vt[64][72];       // [d][key]  (transposed for P@V B-frags)
    __shared__ bf16 Ps[8][16][72];    // per-wave P tile [m][key]

    const int tid  = threadIdx.x;
    const int wave = tid >> 5;
    const int lane = tid & 31;
    const int hi   = lane >> 4;
    const int ln   = lane & 15;

    const int bh = blockIdx.y;
    const int b  = bh >> 4;
    const int h  = bh & 15;
    const int q0 = blockIdx.x * 128 + wave * 16;

    const size_t base = (size_t)bh * SEQ * HD;
    const bf16* Qb = Qh + base;
    const bf16* Kb = Kh + base;
    const bf16* Vb = Vh + base;

    // resident Q A-fragments (16 rows x 64 d = 2 k-steps)
    const bf16* qrow = Qb + (size_t)(q0 + ln) * HD;
    v16bf qf[2];
#pragma unroll
    for (int ks = 0; ks < 2; ++ks)
        qf[ks] = load_a_frag(qrow + ks * 32 + hi * 16);

    v8f oacc[4];
#pragma unroll
    for (int j = 0; j < 4; ++j) oacc[j] = (v8f){};
    float mrow[8], lrow[8];
#pragma unroll
    for (int r = 0; r < 8; ++r) { mrow[r] = -1e30f; lrow[r] = 0.f; }

    const int sr = tid >> 2;         // staging key row 0..63
    const int sc = (tid & 3) * 16;   // staging d col 0,16,32,48

    for (int s0 = 0; s0 < SEQ; s0 += 64) {
        __syncthreads();
        {   // K chunk, row-major: async DMA to LDS (ASYNCcnt path)
            const bf16* src = Kb + (size_t)(s0 + sr) * HD + sc;
            async_cp16(src,     &Ks[sr][sc]);
            async_cp16(src + 8, &Ks[sr][sc + 8]);
            // V chunk, transposed into Vt through VGPRs
            const bf16* vs = Vb + (size_t)(s0 + sr) * HD + sc;
            v8bf v0 = *(const v8bf*)vs;
            v8bf v1 = *(const v8bf*)(vs + 8);
#pragma unroll
            for (int i = 0; i < 8; ++i) {
                Vt[sc + i][sr]     = v0[i];
                Vt[sc + 8 + i][sr] = v1[i];
            }
        }
        wait_async0();
        __syncthreads();

        // ---- scores: S = Q (16x64) x K'^T (64x64) ----
        v8f scs[4];
#pragma unroll
        for (int j = 0; j < 4; ++j) scs[j] = (v8f){};
#pragma unroll
        for (int ks = 0; ks < 2; ++ks) {
#pragma unroll
            for (int j = 0; j < 4; ++j) {
                v16bf kb = load_b_frag(&Ks[j * 16 + ln][ks * 32 + hi * 16]);
                scs[j] = wmma_bf16(qf[ks], kb, scs[j]);
            }
        }
        // padding mask (column = this lane's key for tile j)
#pragma unroll
        for (int j = 0; j < 4; ++j) {
            const bool mk = kpm[(size_t)b * SEQ + s0 + j * 16 + ln] != 0;
#pragma unroll
            for (int r = 0; r < 8; ++r)
                scs[j][r] = mk ? -1e30f : scs[j][r];
        }

        // ---- online softmax (row m = r + 8*hi lives in one 16-lane half) ----
#pragma unroll
        for (int r = 0; r < 8; ++r) {
            float mx = scs[0][r];
#pragma unroll
            for (int j = 1; j < 4; ++j) mx = fmaxf(mx, scs[j][r]);
#pragma unroll
            for (int msk = 8; msk >= 1; msk >>= 1)
                mx = fmaxf(mx, __shfl_xor(mx, msk, 32));
            const float mnew = fmaxf(mrow[r], mx);
            const float corr = __expf(mrow[r] - mnew);
            mrow[r] = mnew;
            float rs = 0.f;
#pragma unroll
            for (int j = 0; j < 4; ++j) {
                const float p = __expf(scs[j][r] - mnew);
                scs[j][r] = p;
                rs += p;
            }
#pragma unroll
            for (int msk = 8; msk >= 1; msk >>= 1)
                rs += __shfl_xor(rs, msk, 32);
            lrow[r] = lrow[r] * corr + rs;
#pragma unroll
            for (int j = 0; j < 4; ++j) oacc[j][r] *= corr;
        }

        // ---- P -> LDS (bf16, A-fragment source layout) ----
#pragma unroll
        for (int j = 0; j < 4; ++j)
#pragma unroll
            for (int r = 0; r < 8; ++r)
                Ps[wave][r + hi * 8][j * 16 + ln] = (bf16)scs[j][r];
        __syncthreads();

        // ---- O += P (16x64) x V (64x64), B-frags from transposed Vt ----
#pragma unroll
        for (int ks = 0; ks < 2; ++ks) {
            v16bf pf = load_a_frag(&Ps[wave][ln][ks * 32 + hi * 16]);
#pragma unroll
            for (int j = 0; j < 4; ++j) {
                v16bf vfr = load_b_frag(&Vt[j * 16 + ln][ks * 32 + hi * 16]);
                oacc[j] = wmma_bf16(pf, vfr, oacc[j]);
            }
        }
    }

    // ---- finalize: O /= l, write [B, T, E] bf16 for the O projection ----
#pragma unroll
    for (int r = 0; r < 8; ++r) {
        const float inv = 1.0f / lrow[r];
        const int m = q0 + r + hi * 8;
#pragma unroll
        for (int j = 0; j < 4; ++j) {
            const int e = h * HD + j * 16 + ln;
            attn[((size_t)b * SEQ + m) * EMBED + e] = (bf16)(oacc[j][r] * inv);
        }
    }
}

// =================================================================
extern "C" void kernel_launch(void* const* d_in, const int* in_sizes, int n_in,
                              void* d_out, int out_size, void* d_ws, size_t ws_size,
                              hipStream_t stream) {
    const float* query = (const float*)d_in[0];
    const float* key   = (const float*)d_in[1];
    const float* value = (const float*)d_in[2];
    const unsigned char* kpm = (const unsigned char*)d_in[3];
    const float* q_w = (const float*)d_in[4];
    const float* q_b = (const float*)d_in[5];
    const float* k_w = (const float*)d_in[6];
    const float* k_b = (const float*)d_in[7];
    const float* v_w = (const float*)d_in[8];
    const float* v_b = (const float*)d_in[9];
    const float* o_w = (const float*)d_in[10];
    const float* o_b = (const float*)d_in[11];
    const float* rel = (const float*)d_in[12];

    const size_t elems = (size_t)BATCH * SEQ * EMBED;  // 4M elements
    bf16* Qh   = (bf16*)d_ws;
    bf16* Kh   = Qh + elems;
    bf16* Vh   = Kh + elems;
    bf16* attn = Vh + elems;   // total 32 MiB of workspace

    dim3 blk(256);
    dim3 gproj(EMBED / 128, (BATCH * SEQ) / 128);  // (8, 32)

    gemm_xwT<float, 0><<<gproj, blk, 0, stream>>>(query, q_w, q_b, nullptr, Qh);
    gemm_xwT<float, 1><<<gproj, blk, 0, stream>>>(key,   k_w, k_b, rel,     Kh);
    gemm_xwT<float, 0><<<gproj, blk, 0, stream>>>(value, v_w, v_b, nullptr, Vh);

    dim3 gattn(SEQ / 128, BATCH * HEADS);          // (8, 64)
    flash_attn<<<gattn, blk, 0, stream>>>(Qh, Kh, Vh, kpm, attn);

    gemm_xwT<bf16, 2><<<gproj, blk, 0, stream>>>(attn, o_w, o_b, nullptr, d_out);
}
